// EfficientKanEnsembleLayer_38293928411732
// MI455X (gfx1250) — matrix operand order repaired and checked by reading, not compiled, not run on hardware
//
#include <hip/hip_runtime.h>
#include <hip/hip_bf16.h>

typedef __attribute__((ext_vector_type(16))) __bf16        v16bf;
typedef __attribute__((ext_vector_type(8)))  float         v8f;
typedef __attribute__((ext_vector_type(4)))  unsigned int  v4u;
typedef __attribute__((ext_vector_type(8)))  int           v8i;
typedef __attribute__((ext_vector_type(4)))  int           v4i;
typedef __attribute__((ext_vector_type(4)))  float         v4f;

#define F_IN   512
#define F_OUT  512
#define GDIM   8
#define KDIM   4608          // 512 features * (1 silu + 8 bases)
#define BM     128
#define BN     128
#define PITCH  288           // per-chunk K width (32 feats * 9), bf16 elements
#define NCHUNK 16            // 512 / 32
#define NROW_TOTAL 32768     // K_ens(8) * B(4096)

union FragBF { v4u u[2]; v16bf v; };

__device__ __forceinline__ unsigned f2bf(float f) {
    unsigned int u = __builtin_bit_cast(unsigned int, f);
    u += 0x7FFFu + ((u >> 16) & 1u);          // round-to-nearest-even
    return u >> 16;
}

// --- Prep: fold base_weight + spline_weight*scaler into one bf16 matrix
// Wc[o][kcol], kcol layout per 32-feature chunk c: [0..31] = base weights,
// [32 + fi*8 + g] = scaled spline weights. Mirrors activation packing.
__global__ __launch_bounds__(256) void kan_prep_kernel(
        const float* __restrict__ bw, const float* __restrict__ sw,
        const float* __restrict__ sc, unsigned short* __restrict__ wc) {
    int idx = blockIdx.x * 256 + threadIdx.x;     // idx = o*512 + f
    int o = idx >> 9, f = idx & 511;
    float b    = bw[idx];
    float scal = sc[idx];
    int   base = o * KDIM + (f >> 5) * PITCH;
    int   fi   = f & 31;
    wc[base + fi] = (unsigned short)f2bf(b);
#pragma unroll
    for (int g = 0; g < GDIM; ++g)
        wc[base + 32 + fi * 8 + g] = (unsigned short)f2bf(sw[idx * GDIM + g] * scal);
}

// --- Main fused kernel: out[n][o] = s * (silu(xr)@Wb^T + bases(xr)@Ws^T)
__global__ __launch_bounds__(256) void kan_main_kernel(
        const float* __restrict__ x, const float* __restrict__ r,
        const float* __restrict__ s, const unsigned short* __restrict__ wc,
        float* __restrict__ out) {
    __shared__ unsigned short As[BM * PITCH];   // activations (silu+bases) bf16
    __shared__ unsigned short Bs[BN * PITCH];   // combined weights bf16

    const int tid   = threadIdx.x;
    const int lane  = tid & 31;
    const int wave  = tid >> 5;                 // 8 waves
    const int waveM = wave >> 2;                // 0..1  (64-row slabs)
    const int waveN = wave & 3;                 // 0..3  (32-col slabs)
    const int mBase = blockIdx.y * BM;
    const int nBase = blockIdx.x * BN;
    const int kens  = mBase >> 12;              // ensemble index (tile within one k)

    v8f acc[4][2];
#pragma unroll
    for (int i = 0; i < 4; ++i)
#pragma unroll
        for (int j = 0; j < 2; ++j) acc[i][j] = (v8f)0.0f;

    const int mrow  = tid >> 1;                 // 0..127: LDS row this thread fills
    const int half  = tid & 1;                  // 16-feature half of the chunk
    const int mfrag = lane & 15;
    const int khalf = lane >> 4;

    // LDS byte offset of Bs (flat->LDS: low 32 bits of generic address, ISA 10.2)
    const unsigned bs_lds = (unsigned)(size_t)&Bs[0];

    for (int c = 0; c < NCHUNK; ++c) {
        // ---- wave 0: kick off TDM DMA of the 128x288 bf16 weight tile ----
        if (wave == 0) {
            unsigned long long ga =
                (unsigned long long)(size_t)(wc + (size_t)nBase * KDIM + c * PITCH);
            v4u g0;
            g0.x = 1u;                                        // count=1, normal mode
            g0.y = bs_lds;                                    // lds_addr
            g0.z = (unsigned)ga;                              // global_addr[31:0]
            g0.w = ((unsigned)(ga >> 32) & 0x01FFFFFFu)       // global_addr[56:32]
                 | (2u << 30);                                // type = 2 ("image")
            v8i g1;
            g1[0] = 0x00010000;                               // data_size=1 (2 bytes)
            g1[1] = (int)((unsigned)(PITCH & 0xFFFF) << 16);  // tensor_dim0 lo16
            g1[2] = (int)((unsigned)(BN & 0xFFFF) << 16);     // td0 hi / tensor_dim1 lo
            g1[3] = (int)((unsigned)PITCH << 16);             // td1 hi / tile_dim0=288
            g1[4] = BN;                                       // tile_dim1=128, tile_dim2=0
            g1[5] = KDIM;                                     // tensor_dim0_stride lo32
            g1[6] = 0;                                        // stride hi / dim1_stride lo
            g1[7] = 0;                                        // dim1_stride hi
            v4i z4 = (v4i)0;
#if __has_include(<hip/amd_detail/amd_gfx1250_TDM.h>)
            v8i z8 = (v8i)0;
            __builtin_amdgcn_tensor_load_to_lds(g0, g1, z4, z4, z8, 0);
#else
            __builtin_amdgcn_tensor_load_to_lds(g0, g1, z4, z4, 0);
#endif
        }

        // ---- all waves: build activation tile (silu + 8 spline bases) ----
        {
            const int f0 = c * 32 + half * 16;
            const float* xp = x + (size_t)(mBase + mrow) * F_IN + f0;
            if (c + 1 < NCHUNK)
                __builtin_prefetch(xp + 32, 0, 0);            // next chunk's x
            float xl[16];
#pragma unroll
            for (int q = 0; q < 4; ++q) {
                v4f t = *(const v4f*)(xp + q * 4);
                xl[q * 4 + 0] = t.x; xl[q * 4 + 1] = t.y;
                xl[q * 4 + 2] = t.z; xl[q * 4 + 3] = t.w;
            }
#pragma unroll
            for (int j = 0; j < 16; ++j) {
                float xr  = xl[j] * r[kens * F_IN + f0 + j];
                float sil = xr * (1.0f / (1.0f + __expf(-xr)));
                int   fi  = half * 16 + j;
                As[mrow * PITCH + fi] = (unsigned short)f2bf(sil);

                // Uniform cubic B-spline, closed form: interval i, frac f.
                // Nonzero bases are slots p..p+3 (p = i-3) with the 4 standard
                // cubic polynomials; everything outside [-2.2, 2.2) is zero.
                float u   = (xr + 2.2f) * 2.5f;               // (x - t0)/h
                float fl  = floorf(u);
                int   i   = (int)fl;
                float f   = u - fl;
                float omf = 1.0f - f;
                float f2  = f * f, f3 = f2 * f;
                const float sixth = 1.0f / 6.0f;
                float w0 = omf * omf * omf * sixth;                            // slot p
                float w1 = (3.0f * f3 - 6.0f * f2 + 4.0f) * sixth;             // p+1
                float w2 = (-3.0f * f3 + 3.0f * f2 + 3.0f * f + 1.0f) * sixth; // p+2
                float w3 = f3 * sixth;                                         // p+3
                unsigned h0 = f2bf(w0), h1 = f2bf(w1), h2 = f2bf(w2), h3 = f2bf(w3);
                // Pair-packed windows for even / odd start alignment
                unsigned E0 = h0 | (h1 << 16), E1 = h2 | (h3 << 16);
                unsigned O0 = h0 << 16, O1 = h1 | (h2 << 16), O2 = h3;
                int  p   = i - 3;
                int  q   = p >> 1;                 // arithmetic shift (neg ok)
                bool odd = (p & 1) != 0;
                unsigned S0 = odd ? O0 : E0;
                unsigned S1 = odd ? O1 : E1;
                unsigned S2 = odd ? O2 : 0u;
                v4u pk;
#pragma unroll
                for (int k = 0; k < 4; ++k) {       // dword k = slots 2k,2k+1
                    unsigned a = (q == k) ? S0 : 0u;
                    a = (q == k - 1) ? S1 : a;
                    a = (q == k - 2) ? S2 : a;
                    pk[k] = a;
                }
                *(v4u*)&As[mrow * PITCH + 32 + fi * 8] = pk;  // 16B-aligned
            }
        }

        // ---- wave 0 drains the DMA, then everyone syncs ----
        if (wave == 0)
            __builtin_amdgcn_s_wait_tensorcnt(0);
        __syncthreads();

        // ---- 9 WMMA K-steps of 32 over the 288-wide chunk ----
#pragma unroll
        for (int kk = 0; kk < 9; ++kk) {
            FragBF bfr[2];
#pragma unroll
            for (int jn = 0; jn < 2; ++jn) {
                // B operand (32x16): lane = column, K = khalf*16 .. +15
                const unsigned short* p =
                    &Bs[(waveN * 32 + jn * 16 + mfrag) * PITCH + kk * 32 + khalf * 16];
                bfr[jn].u[0] = *(const v4u*)p;
                bfr[jn].u[1] = *(const v4u*)(p + 8);
            }
#pragma unroll
            for (int im = 0; im < 4; ++im) {
                // A operand (16x32): lane = row, K = khalf*16 .. +15
                FragBF a;
                const unsigned short* p =
                    &As[(waveM * 64 + im * 16 + mfrag) * PITCH + kk * 32 + khalf * 16];
                a.u[0] = *(const v4u*)p;
                a.u[1] = *(const v4u*)(p + 8);
#pragma unroll
                for (int jn = 0; jn < 2; ++jn)
                    acc[im][jn] = __builtin_amdgcn_wmma_f32_16x16x32_bf16(
                        false, a.v, false, bfr[jn].v, (short)0, acc[im][jn],
                        false, false);
            }
        }
        __syncthreads();
    }

    // ---- epilogue: multiply by sign s[k_ens][o], store f32 ----
#pragma unroll
    for (int jn = 0; jn < 2; ++jn) {
        int   o  = nBase + waveN * 32 + jn * 16 + mfrag;
        float sv = s[kens * F_OUT + o];
#pragma unroll
        for (int im = 0; im < 4; ++im) {
            int gm0 = mBase + waveM * 64 + im * 16 + 8 * khalf;
#pragma unroll
            for (int jr = 0; jr < 8; ++jr)
                out[(size_t)(gm0 + jr) * F_OUT + o] = acc[im][jn][jr] * sv;
        }
    }
}

extern "C" void kernel_launch(void* const* d_in, const int* in_sizes, int n_in,
                              void* d_out, int out_size, void* d_ws, size_t ws_size,
                              hipStream_t stream) {
    const float* x  = (const float*)d_in[0];
    const float* bw = (const float*)d_in[1];
    const float* sw = (const float*)d_in[2];
    const float* sc = (const float*)d_in[3];
    const float* r  = (const float*)d_in[4];
    const float* s  = (const float*)d_in[5];
    unsigned short* wc = (unsigned short*)d_ws;   // 512*4608 bf16 = 4.5 MB

    // 1) fold weights into combined bf16 matrix matching activation packing
    kan_prep_kernel<<<(F_OUT * F_IN) / 256, 256, 0, stream>>>(bw, sw, sc, wc);

    // 2) fused silu+bspline+double-GEMM with TDM weight streaming
    dim3 grid(F_OUT / BN, NROW_TOTAL / BM);       // (4, 256)
    kan_main_kernel<<<grid, 256, 0, stream>>>(x, r, s, wc, (float*)d_out);
}
